// MoELayer_24996709663176
// MI455X (gfx1250) — matrix operand order repaired and checked by reading, not compiled, not run on hardware
//
#include <hip/hip_runtime.h>
#include <hip/hip_bf16.h>
#include <math.h>
#include <stdint.h>

// ---------------------------------------------------------------------------
// MoE (Mixtral-style, top-2 of 8 experts, SwiGLU) for MI455X / gfx1250.
// Sparse grouped-GEMM in bf16 WMMA (v_wmma_f32_16x16x32_bf16, fp32 accum),
// async global->LDS staging (ASYNCcnt) with double buffering in GEMM-2,
// all B-fragments preloaded per K-step so loads overlap WMMA issue.
// ---------------------------------------------------------------------------

typedef __attribute__((ext_vector_type(16))) __bf16 v16bf;
typedef __attribute__((ext_vector_type(8)))  float  v8f;

union FragB { v16bf v; uint4 q[2]; };

constexpr int kT = 8192;   // tokens = B*S
constexpr int kH = 1024;   // hidden
constexpr int kI = 4096;   // intermediate
constexpr int kE = 8;      // experts
constexpr int kK = 2;      // experts per token

// Async global->LDS copy, 16B per lane. lds_off = LDS byte address
// (low 32 bits of generic shared pointer, per ISA aperture mapping).
__device__ __forceinline__ void async_copy_b128(unsigned lds_off, const void* gaddr) {
    asm volatile("global_load_async_to_lds_b128 %0, %1, off"
                 :: "v"(lds_off), "v"(gaddr) : "memory");
}
__device__ __forceinline__ void wait_async0() {
    asm volatile("s_wait_asynccnt 0x0" ::: "memory");
}
__device__ __forceinline__ void wait_async1() {
    asm volatile("s_wait_asynccnt 0x1" ::: "memory");
}

__device__ __forceinline__ void load_frag(FragB& f, const __hip_bfloat16* p) {
    const uint4* q = reinterpret_cast<const uint4*>(p);
    f.q[0] = q[0];
    f.q[1] = q[1];
}

// ---------------------------------------------------------------- utilities
__global__ void zero_f32(float* p, long n4) {
    long i = (long)blockIdx.x * blockDim.x + threadIdx.x;
    float4 z; z.x = z.y = z.z = z.w = 0.f;
    for (; i < n4; i += (long)gridDim.x * blockDim.x)
        reinterpret_cast<float4*>(p)[i] = z;
}

__global__ void zero_i32(int* p, int n) {
    int i = blockIdx.x * blockDim.x + threadIdx.x;
    if (i < n) p[i] = 0;
}

// x fp32 -> bf16, 4 elements per thread
__global__ void cvt_x(const float* __restrict__ x, __hip_bfloat16* __restrict__ xb, long n4) {
    long i = (long)blockIdx.x * blockDim.x + threadIdx.x;
    if (i >= n4) return;
    float4 f = reinterpret_cast<const float4*>(x)[i];
    union { __hip_bfloat16 h[4]; uint2 u; } o;
    o.h[0] = __float2bfloat16(f.x); o.h[1] = __float2bfloat16(f.y);
    o.h[2] = __float2bfloat16(f.z); o.h[3] = __float2bfloat16(f.w);
    reinterpret_cast<uint2*>(xb)[i] = o.u;
}

// src fp32 [R][C] (per expert, blockIdx.z) -> dst bf16 [C][R]  (64x64 LDS tile)
__global__ void transpose_bf16(const float* __restrict__ src, __hip_bfloat16* __restrict__ dst,
                               int R, int C) {
    __shared__ __hip_bfloat16 tile[64 * 68];
    const float* s = src + (size_t)blockIdx.z * R * C;
    __hip_bfloat16* d = dst + (size_t)blockIdx.z * R * C;
    int r0 = blockIdx.y * 64, c0 = blockIdx.x * 64;
    int lane = threadIdx.x & 63, grp = threadIdx.x >> 6;   // 256 threads = 4 row-groups
    for (int rr = grp; rr < 64; rr += 4)
        tile[rr * 68 + lane] = __float2bfloat16(s[(size_t)(r0 + rr) * C + c0 + lane]);
    __syncthreads();
    for (int cc = grp; cc < 64; cc += 4)
        d[(size_t)(c0 + cc) * R + r0 + lane] = tile[lane * 68 + cc];
}

// ------------------------------------------------------------------- gating
// One wave per token: coalesced x reads, shuffle reduction, top-2 routing.
// Softmax denominator cancels under top-k renormalization.
__global__ void gating(const float* __restrict__ x, const float* __restrict__ wg,
                       int* __restrict__ counts, int* __restrict__ slot_token,
                       int* __restrict__ slot_pair, float* __restrict__ pair_w) {
    int wv = threadIdx.x >> 5, lane = threadIdx.x & 31;
    int t = blockIdx.x * 8 + wv;
    const float* xr = x + (size_t)t * kH;
    float acc[kE];
#pragma unroll
    for (int e = 0; e < kE; ++e) acc[e] = 0.f;
    for (int h = lane; h < kH; h += 32) {
        float xv = xr[h];
        const float4* g = reinterpret_cast<const float4*>(wg + (size_t)h * kE);
        float4 g0 = g[0], g1 = g[1];
        acc[0] += xv * g0.x; acc[1] += xv * g0.y; acc[2] += xv * g0.z; acc[3] += xv * g0.w;
        acc[4] += xv * g1.x; acc[5] += xv * g1.y; acc[6] += xv * g1.z; acc[7] += xv * g1.w;
    }
#pragma unroll
    for (int e = 0; e < kE; ++e)
        for (int m = 16; m; m >>= 1) acc[e] += __shfl_xor(acc[e], m, 32);
    if (lane != 0) return;
    float mx = acc[0];
#pragma unroll
    for (int e = 1; e < kE; ++e) mx = fmaxf(mx, acc[e]);
    float pe[kE];
#pragma unroll
    for (int e = 0; e < kE; ++e) pe[e] = __expf(acc[e] - mx);
    int i1 = 0; float b1 = pe[0];
#pragma unroll
    for (int e = 1; e < kE; ++e) if (pe[e] > b1) { b1 = pe[e]; i1 = e; }
    int i2 = (i1 == 0) ? 1 : 0; float b2 = pe[i2];
#pragma unroll
    for (int e = 0; e < kE; ++e) if (e != i1 && pe[e] > b2) { b2 = pe[e]; i2 = e; }
    float rn = 1.f / (b1 + b2);
    int s1 = atomicAdd(&counts[i1], 1);
    slot_token[i1 * kT + s1] = t;
    slot_pair[i1 * kT + s1]  = 2 * t;
    pair_w[2 * t] = b1 * rn;
    int s2 = atomicAdd(&counts[i2], 1);
    slot_token[i2 * kT + s2] = t;
    slot_pair[i2 * kT + s2]  = 2 * t + 1;
    pair_w[2 * t + 1] = b2 * rn;
}

// --------------------------------------------------------- GEMM 1 (hidden)
// hidden[pair, i] = silu(x @ w1) * (x @ w3), grouped per expert.
// Grid: (kI/512, kT/16, kE). Block 256 (8 waves, 64 N-cols each).
__global__ void moe_hidden(const __hip_bfloat16* __restrict__ xb,
                           const __hip_bfloat16* __restrict__ w1T,   // [E][I][H]
                           const __hip_bfloat16* __restrict__ w3T,   // [E][I][H]
                           const int* __restrict__ counts,
                           const int* __restrict__ slot_token,
                           const int* __restrict__ slot_pair,
                           __hip_bfloat16* __restrict__ hidden) {    // [2T][I]
    const int e = blockIdx.z, mt = blockIdx.y, tid = threadIdx.x;
    const int cnt = counts[e];
    if (mt * 16 >= cnt) return;

    __shared__ __hip_bfloat16 xs[16 * 1032];   // 16 token rows, +8 pad
    __shared__ int stok[16], sp[16];
    if (tid < 16) {
        int s = mt * 16 + tid;
        bool v = s < cnt;
        stok[tid] = v ? slot_token[e * kT + s] : -1;
        sp[tid]   = v ? slot_pair[e * kT + s]  : -1;
    }
    __syncthreads();
    // Async-stage 16x1024 bf16 x-tile (gathered rows) into LDS, 16B per op.
    {
        const unsigned xs0 = (unsigned)(uintptr_t)xs;
#pragma unroll
        for (int it = 0; it < 8; ++it) {
            int idx = tid + it * 256;            // 16 rows x 128 16B-chunks
            int r = idx >> 7, c = idx & 127;
            int tok = stok[r]; if (tok < 0) tok = stok[0];   // bounds-safe dummy
            async_copy_b128(xs0 + (unsigned)(r * 1032 + c * 8) * 2,
                            xb + (size_t)tok * kH + c * 8);
        }
        wait_async0();
    }
    __syncthreads();

    const int lane = tid & 31, wv = tid >> 5;
    const int half = lane >> 4, mrow = lane & 15, ncol0 = lane & 15;
    const int nbase = blockIdx.x * 512 + wv * 64;
    const __hip_bfloat16* b1b = w1T + (size_t)e * kI * kH + (size_t)(nbase + ncol0) * kH;
    const __hip_bfloat16* b3b = w3T + (size_t)e * kI * kH + (size_t)(nbase + ncol0) * kH;

    v8f z{};
    v8f acc1[4], acc3[4];
#pragma unroll
    for (int j = 0; j < 4; ++j) { acc1[j] = z; acc3[j] = z; }

    for (int kc = 0; kc < kH; kc += 32) {
        FragB a;   // A layout: e0..15 -> K=kc+16*half+e (matches B)
        a.q[0] = *reinterpret_cast<const uint4*>(xs + mrow * 1032 + kc + 16 * half);
        a.q[1] = *reinterpret_cast<const uint4*>(xs + mrow * 1032 + kc + 16 * half + 8);
        const int kb = kc + 16 * half;   // B: 16 contiguous K per lane-half
        if (kc + 32 < kH) {              // next K-slice into nearer caches
            __builtin_prefetch(b1b + kb + 32, 0, 3);
            __builtin_prefetch(b3b + kb + 32, 0, 3);
        }
        // All 8 B-fragments simultaneously live: one load clause, then the
        // WMMA chain drains it with partial loadcnt waits (load/WMMA overlap).
        FragB b1f[4], b3f[4];
#pragma unroll
        for (int j = 0; j < 4; ++j) {
            load_frag(b1f[j], b1b + (size_t)j * 16 * kH + kb);
            load_frag(b3f[j], b3b + (size_t)j * 16 * kH + kb);
        }
#pragma unroll
        for (int j = 0; j < 4; ++j) {
            acc1[j] = __builtin_amdgcn_wmma_f32_16x16x32_bf16(
                false, a.v, false, b1f[j].v, (short)0, acc1[j], false, false);
            acc3[j] = __builtin_amdgcn_wmma_f32_16x16x32_bf16(
                false, a.v, false, b3f[j].v, (short)0, acc3[j], false, false);
        }
    }
    // Fused SwiGLU epilogue: hidden = silu(g1) * g3, bf16 store.
#pragma unroll
    for (int j = 0; j < 4; ++j) {
        const int n = nbase + j * 16 + ncol0;
#pragma unroll
        for (int v = 0; v < 8; ++v) {
            const int r = v + 8 * half;      // C/D layout: row = vgpr + 8*(lane/16)
            const int p = sp[r];
            if (p >= 0) {
                float g = acc1[j][v];
                float hv = (g / (1.f + __expf(-g))) * acc3[j][v];
                hidden[(size_t)p * kI + n] = __float2bfloat16(hv);
            }
        }
    }
}

// ------------------------------------------------------------ GEMM 2 (out)
// out[t] += pair_w * (hidden @ w2), grouped per expert.
// K = kI streamed in 128-wide chunks, double-buffered async LDS staging.
// Grid: (kH/512, kT/16, kE). Block 256.
__global__ void moe_out(const __hip_bfloat16* __restrict__ hidden,   // [2T][I]
                        const __hip_bfloat16* __restrict__ w2T,      // [E][H][I]
                        const int* __restrict__ counts,
                        const int* __restrict__ slot_token,
                        const int* __restrict__ slot_pair,
                        const float* __restrict__ pair_w,
                        float* __restrict__ out) {
    const int e = blockIdx.z, mt = blockIdx.y, tid = threadIdx.x;
    const int cnt = counts[e];
    if (mt * 16 >= cnt) return;

    __shared__ __hip_bfloat16 hs[2][16 * 136];  // double-buffered 16x128 K-chunks
    __shared__ int stok[16], sp[16];
    __shared__ float swt[16];
    if (tid < 16) {
        int s = mt * 16 + tid;
        bool v = s < cnt;
        int p = v ? slot_pair[e * kT + s] : -1;
        stok[tid] = v ? slot_token[e * kT + s] : -1;
        sp[tid] = p;
        swt[tid] = (p >= 0) ? pair_w[p] : 0.f;
    }
    __syncthreads();

    const int lane = tid & 31, wv = tid >> 5;
    const int half = lane >> 4, mrow = lane & 15, ncol0 = lane & 15;
    const int nbase = blockIdx.x * 512 + wv * 64;
    const __hip_bfloat16* b2b = w2T + (size_t)e * kH * kI + (size_t)(nbase + ncol0) * kI;
    const unsigned hs0 = (unsigned)(uintptr_t)&hs[0][0];

    // Per-stage: 256 threads x 16B = one async b128 op per thread (1/wave).
    const int srow = tid >> 4, scol = tid & 15;
    int sprow = sp[srow]; if (sprow < 0) sprow = sp[0];    // bounds-safe dummy
    const __hip_bfloat16* srcrow = hidden + (size_t)sprow * kI + scol * 8;
    const unsigned ldsoff = (unsigned)(srow * 136 + scol * 8) * 2;

    v8f z{};
    v8f acc[4];
#pragma unroll
    for (int j = 0; j < 4; ++j) acc[j] = z;

    async_copy_b128(hs0 + ldsoff, srcrow);                 // prologue: chunk 0
    constexpr int NCHUNK = kI / 128;
    for (int ch = 0; ch < NCHUNK; ++ch) {
        const int cur = ch & 1;
        if (ch + 1 < NCHUNK) {
            async_copy_b128(hs0 + (unsigned)((cur ^ 1) * 16 * 136 * 2) + ldsoff,
                            srcrow + (ch + 1) * 128);
            wait_async1();                                  // chunk ch resident
        } else {
            wait_async0();
        }
        __syncthreads();
        const __hip_bfloat16* hb = hs[cur];
        const int kk = ch * 128;
#pragma unroll
        for (int kc2 = 0; kc2 < 128; kc2 += 32) {
            FragB a;
            a.q[0] = *reinterpret_cast<const uint4*>(hb + mrow * 136 + kc2 + 16 * half);
            a.q[1] = *reinterpret_cast<const uint4*>(hb + mrow * 136 + kc2 + 16 * half + 8);
            const int kb = kk + kc2 + 16 * half;
            if (ch + 1 < NCHUNK)                            // next w2T K-chunk
                __builtin_prefetch(b2b + kb + 128, 0, 3);
            FragB bf[4];
#pragma unroll
            for (int j = 0; j < 4; ++j)
                load_frag(bf[j], b2b + (size_t)j * 16 * kI + kb);
#pragma unroll
            for (int j = 0; j < 4; ++j)
                acc[j] = __builtin_amdgcn_wmma_f32_16x16x32_bf16(
                    false, a.v, false, bf[j].v, (short)0, acc[j], false, false);
        }
        __syncthreads();
    }
    // Scale by routing weight; exactly 2 commutative atomic adds per out elem.
#pragma unroll
    for (int j = 0; j < 4; ++j) {
        const int n = nbase + j * 16 + ncol0;
#pragma unroll
        for (int v = 0; v < 8; ++v) {
            const int r = v + 8 * half;
            if (sp[r] >= 0)
                atomicAdd(out + (size_t)stok[r] * kH + n, acc[j][v] * swt[r]);
        }
    }
}

// ------------------------------------------------------------------ launch
extern "C" void kernel_launch(void* const* d_in, const int* in_sizes, int n_in,
                              void* d_out, int out_size, void* d_ws, size_t ws_size,
                              hipStream_t stream) {
    const float* x      = (const float*)d_in[0];
    const float* w_gate = (const float*)d_in[1];
    const float* w1     = (const float*)d_in[2];
    const float* w2     = (const float*)d_in[3];
    const float* w3     = (const float*)d_in[4];
    float* out = (float*)d_out;
    (void)in_sizes; (void)n_in; (void)out_size; (void)ws_size;

    char* ws = (char*)d_ws;
    size_t off = 0;
    auto take = [&](size_t bytes) { char* p = ws + off; off = (off + bytes + 255) & ~(size_t)255; return p; };
    int*            counts     = (int*)take(kE * sizeof(int));
    int*            slot_token = (int*)take((size_t)kE * kT * sizeof(int));
    int*            slot_pair  = (int*)take((size_t)kE * kT * sizeof(int));
    float*          pair_w     = (float*)take((size_t)kK * kT * sizeof(float));
    __hip_bfloat16* xb         = (__hip_bfloat16*)take((size_t)kT * kH * 2);
    __hip_bfloat16* w1T        = (__hip_bfloat16*)take((size_t)kE * kH * kI * 2);
    __hip_bfloat16* w3T        = (__hip_bfloat16*)take((size_t)kE * kH * kI * 2);
    __hip_bfloat16* w2T        = (__hip_bfloat16*)take((size_t)kE * kH * kI * 2);
    __hip_bfloat16* hidden     = (__hip_bfloat16*)take((size_t)kK * kT * kI * 2);

    // 0) init
    zero_i32<<<1, 32, 0, stream>>>(counts, kE);
    zero_f32<<<2048, 256, 0, stream>>>(out, (long)kT * kH / 4);
    // 1) precision conversion / weight transposition (bf16, K-contiguous)
    cvt_x<<<(kT * kH / 4 + 255) / 256, 256, 0, stream>>>(x, xb, (long)kT * kH / 4);
    transpose_bf16<<<dim3(kI / 64, kH / 64, kE), 256, 0, stream>>>(w1, w1T, kH, kI);
    transpose_bf16<<<dim3(kI / 64, kH / 64, kE), 256, 0, stream>>>(w3, w3T, kH, kI);
    transpose_bf16<<<dim3(kH / 64, kI / 64, kE), 256, 0, stream>>>(w2, w2T, kI, kH);
    // 2) routing
    gating<<<kT / 8, 256, 0, stream>>>(x, w_gate, counts, slot_token, slot_pair, pair_w);
    // 3) grouped SwiGLU GEMMs
    moe_hidden<<<dim3(kI / 512, kT / 16, kE), 256, 0, stream>>>(
        xb, w1T, w3T, counts, slot_token, slot_pair, hidden);
    moe_out<<<dim3(kH / 512, kT / 16, kE), 256, 0, stream>>>(
        hidden, w2T, counts, slot_token, slot_pair, pair_w, out);
}
